// PreimageEnergyDecoder_29102698397752
// MI455X (gfx1250) — matrix operand order repaired and checked by reading, not compiled, not run on hardware
//
#include <hip/hip_runtime.h>

// Problem constants (match reference)
#define BB 32768
#define DD 256
#define YDD 64
#define TTK 32      // knots / buckets
#define RRK 16      // low-rank dim
#define HH 256
#define NSTEPS 8
#define NTMAX 2080  // max 16-row tiles: 32768/16 + 32 buckets padding

typedef __attribute__((ext_vector_type(16))) __bf16 v16bf;
typedef __attribute__((ext_vector_type(8)))  __bf16 v8bf;
typedef __attribute__((ext_vector_type(8)))  float  v8f;
typedef unsigned short u16;
typedef unsigned int   u32;

// ---------- helpers ----------
__device__ __forceinline__ u16 f2bf(float f) {      // RNE float->bf16 bits
  u32 u = __builtin_bit_cast(u32, f);
  u32 r = u + 0x7FFFu + ((u >> 16) & 1u);
  return (u16)(r >> 16);
}
__device__ __forceinline__ float softplusf(float x) {
  return x > 20.f ? x : log1pf(__expf(x));
}
// branchless SELU on fast exp (no exec-mask divergence)
__device__ __forceinline__ float seluf(float x) {
  const float sc = 1.0507009873554805f, scal = 1.7580993408473766f; // sc*alpha
  float e = __expf(fminf(x, 0.f));
  float neg = scal * (e - 1.f);
  return x > 0.f ? sc * x : neg;
}
__device__ __forceinline__ v8f wmma_bf16(v16bf a, v16bf b, v8f c) {
  return __builtin_amdgcn_wmma_f32_16x16x32_bf16(false, a, false, b, (short)0, c, false, false);
}
__device__ __forceinline__ v8f vzero8() {
  v8f z;
#pragma unroll
  for (int i = 0; i < 8; i++) z[i] = 0.f;
  return z;
}
// A fragment of a row-major 16xK tile (stride in elems), K-chunk at kbase (mult of 32)
__device__ __forceinline__ v16bf loadA(const u16* base, int stride, int kbase, int lane15, int hi) {
  const __bf16* p = (const __bf16*)(base + lane15 * stride + kbase + hi * 8);
  v8bf lo = *(const v8bf*)p;
  v8bf hv = *(const v8bf*)(p + 16);
  v16bf r;
#pragma unroll
  for (int i = 0; i < 8; i++) { r[i] = lo[i]; r[i + 8] = hv[i]; }
  return r;
}
// B fragment from a [N][K] array (stride K): lane reads K-values hi*8..+7 and 16+hi*8..+7 of its column
__device__ __forceinline__ v16bf loadB(const u16* base, int stride, int nbase, int kbase, int lane15, int hi) {
  const __bf16* p = (const __bf16*)(base + (nbase + lane15) * stride + kbase + hi * 8);
  v8bf lo = *(const v8bf*)p;
  v8bf hv = *(const v8bf*)(p + 16);
  v16bf r;
#pragma unroll
  for (int i = 0; i < 8; i++) { r[i] = lo[i]; r[i + 8] = hv[i]; }
  return r;
}

// ---------- prep kernels ----------
__global__ void k_zero(int* cnt) { if (threadIdx.x < TTK) cnt[threadIdx.x] = 0; }

__global__ void k_bin(const float* __restrict__ t, const float* __restrict__ knots,
                      int* __restrict__ idxarr, int* __restrict__ rankarr, int* __restrict__ cnt) {
  __shared__ float sk[TTK];
  int tid = threadIdx.x;
  if (tid < TTK) sk[tid] = knots[tid];
  __syncthreads();
  int b = blockIdx.x * blockDim.x + tid;
  if (b >= BB) return;
  float tt = t[b];
  int raw = 0;
#pragma unroll
  for (int i = 0; i < TTK; i++) raw += (sk[i] < tt) ? 1 : 0;   // searchsorted left
  int idx1 = min(max(raw, 1), TTK - 1);
  int idx0 = idx1 - 1;
  float w = (tt - sk[idx0]) / (sk[idx1] - sk[idx0]);
  int idx = (w >= 0.5f) ? idx1 : idx0;
  idxarr[b] = idx;
  rankarr[b] = atomicAdd(&cnt[idx], 1);
}

__global__ void k_prefix(const int* __restrict__ cnt, int* __restrict__ rawoff,
                         int* __restrict__ tb, int* __restrict__ tbase) {
  int off = 0;
  for (int i = 0; i < TTK; i++) { rawoff[i] = off; off += cnt[i]; }
  int nt = 0;
  for (int i = 0; i < TTK; i++)
    for (int base = 0; base < cnt[i]; base += 16) { tb[nt] = i; tbase[nt] = base; nt++; }
  for (; nt < NTMAX; nt++) { tb[nt] = -1; tbase[nt] = 0; }
}

__global__ void k_scatter(const int* __restrict__ idxarr, const int* __restrict__ rankarr,
                          const int* __restrict__ rawoff, int* __restrict__ members) {
  int b = blockIdx.x * blockDim.x + threadIdx.x;
  if (b >= BB) return;
  members[rawoff[idxarr[b]] + rankarr[b]] = b;
}

// dst[n*K+k] = bf16(src[k*N+n])  (weight transpose into [N][K] bf16)
__global__ void k_tr(const float* __restrict__ src, u16* __restrict__ dst, int K, int N) {
  int i = blockIdx.x * blockDim.x + threadIdx.x;
  if (i >= K * N) return;
  int n = i / K, k = i % K;
  dst[n * K + k] = f2bf(src[k * N + n]);
}
__global__ void k_cvt(const float* __restrict__ src, u16* __restrict__ dst, int n) {
  int i = blockIdx.x * blockDim.x + threadIdx.x;
  if (i < n) dst[i] = f2bf(src[i]);
}

// ---------- main kernel: one 16-row bucket-uniform tile per block, 4 waves ----------
__launch_bounds__(128)
__global__ void k_main(
    const float* __restrict__ y, const float* __restrict__ t,
    const float* __restrict__ alpha_raw, const float* __restrict__ d_raw,
    const float* __restrict__ Ug, const float* __restrict__ prior_mu, const float* __restrict__ prior_sigma,
    const float* __restrict__ b_enc, const float* __restrict__ b_in, const float* __restrict__ b_h1,
    const float* __restrict__ b_h2, const float* __restrict__ b_out,
    const u16* __restrict__ Bs1,   // W_enc as-is    [64][256]  (B for r = x @ Wenc^T)
    const u16* __restrict__ Bs2,   // W_enc^T        [256][64]  (B for g = r @ Wenc)
    const u16* __restrict__ WinT,  // [256][96]
    const u16* __restrict__ Wh1T, const u16* __restrict__ Wh2T, const u16* __restrict__ WoutT, // [256][256]
    const int* __restrict__ tile_bucket, const int* __restrict__ tile_base,
    const int* __restrict__ cnt, const int* __restrict__ rawoff, const int* __restrict__ members,
    float* __restrict__ out) {
  int tid = threadIdx.x;
  int bucket = tile_bucket[blockIdx.x];
  if (bucket < 0) return;                 // uniform per block: safe before barriers
  int base = tile_base[blockIdx.x];

  __shared__ u16   sA0[16 * 96];          // MLP input (y | temb), bf16
  __shared__ float sY[16 * 64];
  __shared__ u16   sActA[16 * 256];       // ping (also xs/gs in step loop)
  __shared__ u16   sActB[16 * 256];       // pong
  __shared__ u16   sRs[16 * 64];
  __shared__ u16   sProjA[16 * 32];       // K padded 16->32 with zeros
  __shared__ u16   sUt[16 * 256];         // U^T  [r][d]  (B for proj)
  __shared__ u16   sUpg[256 * 32];        // U    [d][r..pad] (B for pg)
  __shared__ float sDiag[256], sMu[256], sIv[256];
  __shared__ float sAlpha;
  __shared__ int   sSid[16], sActive[16];

  int bcnt = cnt[bucket], roff = rawoff[bucket];
  if (tid < 16) {
    int rank = base + tid;
    int act = rank < bcnt;
    sSid[tid] = members[roff + (act ? rank : (bcnt - 1))]; // dup last row -> EXEC stays full
    sActive[tid] = act;
  }
  if (tid == 0) sAlpha = softplusf(alpha_raw[bucket]);
  __syncthreads();

  // stage y (fp32 + bf16 into MLP input)
  for (int i = tid; i < 16 * 64; i += 128) {
    int m = i >> 6, c = i & 63;
    float v = y[sSid[m] * YDD + c];
    sY[m * 64 + c] = v;
    sA0[m * 96 + c] = f2bf(v);
  }
  // time embedding: [sin(t*f_j), cos(t*f_j)], f_j = exp(-ln(1e4)*j/16)
  for (int i = tid; i < 256; i += 128) {
    int m = i >> 4, j = i & 15;
    float tv = t[sSid[m]];
    float fr = __expf(-logf(10000.f) * (float)j / 16.f);
    float a = tv * fr;
    sA0[m * 96 + 64 + j] = f2bf(__sinf(a));
    sA0[m * 96 + 80 + j] = f2bf(__cosf(a));
  }
  // per-bucket column params
  for (int i = tid; i < 256; i += 128) {
    float d = softplusf(d_raw[bucket * DD + i]);
    sDiag[i] = d * d + 1e-4f;
    sMu[i] = prior_mu[bucket * DD + i];
    float sg = fmaxf(prior_sigma[bucket * DD + i], 1e-6f);
    sIv[i] = 0.01f / (sg * sg);
  }
  // U in both layouts (bf16); pad pg-B K to 32 with zeros
  for (int i = tid; i < DD * RRK; i += 128) {
    int dd = i >> 4, r = i & 15;
    u16 ub = f2bf(Ug[(bucket * DD + dd) * RRK + r]);
    sUt[r * 256 + dd] = ub;
    sUpg[dd * 32 + r] = ub;
    sUpg[dd * 32 + 16 + r] = 0;
  }
  for (int i = tid; i < 16 * 16; i += 128)   // zero upper K-half of proj A once
    sProjA[(i >> 4) * 32 + 16 + (i & 15)] = 0;
  __syncthreads();

  int lane = tid & 31, wv = tid >> 5;
  int lane15 = lane & 15, hi = lane >> 4;
  int nb[4];
  float dgv[4], muv[4], ivv[4], biv[4], bh1v[4], bh2v[4], bov[4];
#pragma unroll
  for (int j = 0; j < 4; j++) {
    nb[j] = wv * 64 + j * 16;
    int col = nb[j] + lane15;
    dgv[j] = sDiag[col]; muv[j] = sMu[col]; ivv[j] = sIv[col];
    biv[j] = b_in[col]; bh1v[j] = b_h1[col]; bh2v[j] = b_h2[col]; bov[j] = b_out[col];
  }
  float bencv = b_enc[wv * 16 + lane15];
  float alpha = sAlpha;

  v8f xc[4];
  // ---------------- MLP warm start ----------------
  {
    v8f acc[4];
    // layer 1: A = sA0 (K=96) -> sActA
    {
      v16bf a0[3];
#pragma unroll
      for (int kc = 0; kc < 3; kc++) a0[kc] = loadA(sA0, 96, kc * 32, lane15, hi);
#pragma unroll
      for (int j = 0; j < 4; j++) acc[j] = vzero8();
#pragma unroll
      for (int kc = 0; kc < 3; kc++)
#pragma unroll
        for (int j = 0; j < 4; j++)
          acc[j] = wmma_bf16(a0[kc], loadB(WinT, 96, nb[j], kc * 32, lane15, hi), acc[j]);
#pragma unroll
      for (int j = 0; j < 4; j++)
#pragma unroll
        for (int i = 0; i < 8; i++)
          sActA[(i + hi * 8) * 256 + nb[j] + lane15] = f2bf(seluf(acc[j][i] + biv[j]));
    }
    __syncthreads();

    v16bf af[8];
    // layer 2: sActA -> sActB
#pragma unroll
    for (int kc = 0; kc < 8; kc++) af[kc] = loadA(sActA, 256, kc * 32, lane15, hi);
#pragma unroll
    for (int j = 0; j < 4; j++) acc[j] = vzero8();
#pragma unroll
    for (int kc = 0; kc < 8; kc++)
#pragma unroll
      for (int j = 0; j < 4; j++)
        acc[j] = wmma_bf16(af[kc], loadB(Wh1T, 256, nb[j], kc * 32, lane15, hi), acc[j]);
#pragma unroll
    for (int j = 0; j < 4; j++)
#pragma unroll
      for (int i = 0; i < 8; i++)
        sActB[(i + hi * 8) * 256 + nb[j] + lane15] = f2bf(seluf(acc[j][i] + bh1v[j]));
    __syncthreads();

    // layer 3: sActB -> sActA
#pragma unroll
    for (int kc = 0; kc < 8; kc++) af[kc] = loadA(sActB, 256, kc * 32, lane15, hi);
#pragma unroll
    for (int j = 0; j < 4; j++) acc[j] = vzero8();
#pragma unroll
    for (int kc = 0; kc < 8; kc++)
#pragma unroll
      for (int j = 0; j < 4; j++)
        acc[j] = wmma_bf16(af[kc], loadB(Wh2T, 256, nb[j], kc * 32, lane15, hi), acc[j]);
#pragma unroll
    for (int j = 0; j < 4; j++)
#pragma unroll
      for (int i = 0; i < 8; i++)
        sActA[(i + hi * 8) * 256 + nb[j] + lane15] = f2bf(seluf(acc[j][i] + bh2v[j]));
    __syncthreads();

    // output layer: sActA -> xc (registers)
#pragma unroll
    for (int kc = 0; kc < 8; kc++) af[kc] = loadA(sActA, 256, kc * 32, lane15, hi);
#pragma unroll
    for (int j = 0; j < 4; j++) acc[j] = vzero8();
#pragma unroll
    for (int kc = 0; kc < 8; kc++)
#pragma unroll
      for (int j = 0; j < 4; j++)
        acc[j] = wmma_bf16(af[kc], loadB(WoutT, 256, nb[j], kc * 32, lane15, hi), acc[j]);
#pragma unroll
    for (int j = 0; j < 4; j++)
#pragma unroll
      for (int i = 0; i < 8; i++) xc[j][i] = acc[j][i] + bov[j];
    __syncthreads();   // done reading sActA before step loop rewrites it
  }

  // hoist loop-invariant B fragments for the pg GEMM (U in LDS, K padded to 32)
  v16bf bupg[4];
#pragma unroll
  for (int j = 0; j < 4; j++) bupg[j] = loadB(sUpg, 32, nb[j], 0, lane15, hi);

  // ---------------- 8 preconditioned gradient steps ----------------
  for (int s = 0; s < NSTEPS; s++) {
    // (1) x -> LDS bf16 (A layout source)
#pragma unroll
    for (int j = 0; j < 4; j++)
#pragma unroll
      for (int i = 0; i < 8; i++)
        sActA[(i + hi * 8) * 256 + nb[j] + lane15] = f2bf(xc[j][i]);
    __syncthreads();

    // (2) r = x @ Wenc^T + b_enc - y  (each wave: 16 of 64 cols; 2 independent chains)
    {
      v8f r0 = vzero8(), r1 = vzero8();
#pragma unroll
      for (int kc = 0; kc < 8; kc += 2) {
        r0 = wmma_bf16(loadA(sActA, 256, kc * 32, lane15, hi),
                       loadB(Bs1, 256, wv * 16, kc * 32, lane15, hi), r0);
        r1 = wmma_bf16(loadA(sActA, 256, (kc + 1) * 32, lane15, hi),
                       loadB(Bs1, 256, wv * 16, (kc + 1) * 32, lane15, hi), r1);
      }
      int colr = wv * 16 + lane15;
#pragma unroll
      for (int i = 0; i < 8; i++) {
        float rv = r0[i] + r1[i] + bencv - sY[(i + hi * 8) * 64 + colr];
        sRs[(i + hi * 8) * 64 + colr] = f2bf(rv);
      }
    }
    __syncthreads();

    // (3) g = r @ Wenc + iv*(x - mu)   (4 independent accumulators)
    v8f gg[4];
    {
      v16bf ar[2];
#pragma unroll
      for (int kc = 0; kc < 2; kc++) ar[kc] = loadA(sRs, 64, kc * 32, lane15, hi);
#pragma unroll
      for (int j = 0; j < 4; j++) gg[j] = vzero8();
#pragma unroll
      for (int kc = 0; kc < 2; kc++)
#pragma unroll
        for (int j = 0; j < 4; j++)
          gg[j] = wmma_bf16(ar[kc], loadB(Bs2, 64, nb[j], kc * 32, lane15, hi), gg[j]);
#pragma unroll
      for (int j = 0; j < 4; j++)
#pragma unroll
        for (int i = 0; i < 8; i++) {
          gg[j][i] += ivv[j] * (xc[j][i] - muv[j]);
          sActA[(i + hi * 8) * 256 + nb[j] + lane15] = f2bf(gg[j][i]);
        }
    }
    __syncthreads();

    // (4) proj = g @ U  (16x16; all waves compute, wave0 publishes; 2 chains)
    {
      v8f p0 = vzero8(), p1 = vzero8();
#pragma unroll
      for (int kc = 0; kc < 8; kc += 2) {
        p0 = wmma_bf16(loadA(sActA, 256, kc * 32, lane15, hi),
                       loadB(sUt, 256, 0, kc * 32, lane15, hi), p0);
        p1 = wmma_bf16(loadA(sActA, 256, (kc + 1) * 32, lane15, hi),
                       loadB(sUt, 256, 0, (kc + 1) * 32, lane15, hi), p1);
      }
      if (wv == 0)
#pragma unroll
        for (int i = 0; i < 8; i++) sProjA[(i + hi * 8) * 32 + lane15] = f2bf(p0[i] + p1[i]);
    }
    __syncthreads();

    // (5) pg = diag*g + proj @ U^T ;  x -= alpha*pg  (4 independent WMMAs, K=32)
    {
      v16bf ap = loadA(sProjA, 32, 0, lane15, hi);
      v8f acc[4];
#pragma unroll
      for (int j = 0; j < 4; j++) acc[j] = vzero8();
#pragma unroll
      for (int j = 0; j < 4; j++) acc[j] = wmma_bf16(ap, bupg[j], acc[j]);
#pragma unroll
      for (int j = 0; j < 4; j++)
#pragma unroll
        for (int i = 0; i < 8; i++)
          xc[j][i] -= alpha * (dgv[j] * gg[j][i] + acc[j][i]);
    }
    // no barrier needed: next write to sActA is fenced by the barrier after step (4)
  }

  // write out active rows
#pragma unroll
  for (int j = 0; j < 4; j++)
#pragma unroll
    for (int i = 0; i < 8; i++) {
      int m = i + hi * 8;
      if (sActive[m]) out[sSid[m] * DD + nb[j] + lane15] = xc[j][i];
    }
}

// ---------- host launcher ----------
extern "C" void kernel_launch(void* const* d_in, const int* in_sizes, int n_in,
                              void* d_out, int out_size, void* d_ws, size_t ws_size,
                              hipStream_t stream) {
  const float* y        = (const float*)d_in[0];
  const float* t        = (const float*)d_in[1];
  const float* t_knots  = (const float*)d_in[2];
  const float* alpha_r  = (const float*)d_in[3];
  const float* d_raw    = (const float*)d_in[4];
  const float* U        = (const float*)d_in[5];
  const float* pmu      = (const float*)d_in[6];
  const float* psig     = (const float*)d_in[7];
  const float* W_enc    = (const float*)d_in[8];
  const float* b_enc    = (const float*)d_in[9];
  const float* W_in     = (const float*)d_in[10];
  const float* b_in     = (const float*)d_in[11];
  const float* W_h1     = (const float*)d_in[12];
  const float* b_h1     = (const float*)d_in[13];
  const float* W_h2     = (const float*)d_in[14];
  const float* b_h2     = (const float*)d_in[15];
  const float* W_out    = (const float*)d_in[16];
  const float* b_out    = (const float*)d_in[17];
  float* out = (float*)d_out;

  char* ws = (char*)d_ws;
  size_t off = 0;
  auto take = [&](size_t bytes) -> void* {
    void* p = ws + off;
    off = (off + bytes + 255) & ~(size_t)255;
    return p;
  };
  int* cnt     = (int*)take(TTK * 4);
  int* rawoff  = (int*)take(TTK * 4);
  int* tb      = (int*)take(NTMAX * 4);
  int* tbase   = (int*)take(NTMAX * 4);
  int* idxarr  = (int*)take(BB * 4);
  int* rankarr = (int*)take(BB * 4);
  int* members = (int*)take(BB * 4);
  u16* Bs1   = (u16*)take((size_t)YDD * DD * 2);
  u16* Bs2   = (u16*)take((size_t)DD * YDD * 2);
  u16* WinT  = (u16*)take((size_t)HH * 96 * 2);
  u16* Wh1T  = (u16*)take((size_t)HH * HH * 2);
  u16* Wh2T  = (u16*)take((size_t)HH * HH * 2);
  u16* WoutT = (u16*)take((size_t)DD * HH * 2);

  k_zero<<<1, 32, 0, stream>>>(cnt);
  k_bin<<<BB / 256, 256, 0, stream>>>(t, t_knots, idxarr, rankarr, cnt);
  k_prefix<<<1, 1, 0, stream>>>(cnt, rawoff, tb, tbase);
  k_scatter<<<BB / 256, 256, 0, stream>>>(idxarr, rankarr, rawoff, members);

  k_tr<<<(96 * 256 + 255) / 256, 256, 0, stream>>>(W_in, WinT, 96, 256);     // [256][96]
  k_tr<<<(256 * 256 + 255) / 256, 256, 0, stream>>>(W_h1, Wh1T, 256, 256);
  k_tr<<<(256 * 256 + 255) / 256, 256, 0, stream>>>(W_h2, Wh2T, 256, 256);
  k_tr<<<(256 * 256 + 255) / 256, 256, 0, stream>>>(W_out, WoutT, 256, 256);
  k_tr<<<(64 * 256 + 255) / 256, 256, 0, stream>>>(W_enc, Bs2, 64, 256);     // Wenc^T [256][64]
  k_cvt<<<(64 * 256 + 255) / 256, 256, 0, stream>>>(W_enc, Bs1, 64 * 256);   // Wenc   [64][256]

  k_main<<<NTMAX, 128, 0, stream>>>(
      y, t, alpha_r, d_raw, U, pmu, psig,
      b_enc, b_in, b_h1, b_h2, b_out,
      Bs1, Bs2, WinT, Wh1T, Wh2T, WoutT,
      tb, tbase, cnt, rawoff, members, out);
}
